// GTLayer_28827820490818
// MI455X (gfx1250) — matrix operand order repaired and checked by reading, not compiled, not run on hardware
//
#include <hip/hip_runtime.h>
#include <hip/hip_bf16.h>

// ---------------------------------------------------------------------------
// Problem constants (match the reference)
// ---------------------------------------------------------------------------
constexpr int NB  = 2;     // batch
constexpr int NL  = 2048;  // sequence length
constexpr int ND  = 1024;  // model dim
constexpr int NH  = 8;     // heads
constexpr int NHD = 128;   // head dim
constexpr int NRL = 32;    // relative dim

typedef _Float16 half_t;
typedef __attribute__((ext_vector_type(16))) _Float16 v16h;
typedef __attribute__((ext_vector_type(8)))  _Float16 h8;
typedef __attribute__((ext_vector_type(8)))  float    v8f;
typedef __attribute__((ext_vector_type(4)))  int      v4i;

// CDNA5 async global->LDS path (existence probe-confirmed by round-2 error).
#if defined(__has_builtin)
#  if __has_builtin(__builtin_amdgcn_global_load_async_to_lds_b128)
#    define HAVE_ASYNC_LDS 1
#  endif
#  if __has_builtin(__builtin_amdgcn_s_wait_asynccnt)
#    define HAVE_WAIT_ASYNC 1
#  endif
#endif

__device__ inline void wait_async_lds() {
#if defined(HAVE_ASYNC_LDS)
#  if defined(HAVE_WAIT_ASYNC)
  __builtin_amdgcn_s_wait_asynccnt(0);
#  else
  asm volatile("s_wait_asynccnt 0x0" ::: "memory");
#  endif
#endif
}

// One 16B async DMA chunk: global (AS1) -> LDS (AS3), tracked by ASYNCcnt.
__device__ inline void async_copy_b128(const half_t* src, half_t* dst) {
#if defined(HAVE_ASYNC_LDS)
  v4i* s0 = (v4i*)src;  // drop const / re-pointee (generic)
  __builtin_amdgcn_global_load_async_to_lds_b128(
      (__attribute__((address_space(1))) v4i*)s0,
      (__attribute__((address_space(3))) v4i*)dst,
      0, 0);
#else
  *(h8*)dst = *(const h8*)src;
#endif
}

union HFrag { v16h v; h8 q[2]; };

__device__ inline v8f vzero8() {
  v8f z = {0.f, 0.f, 0.f, 0.f, 0.f, 0.f, 0.f, 0.f};
  return z;
}

// A-fragment (16xK row-major tile): lane row = lane&15,
// half i -> K = (i&7) + (i>>3)*16 + g*8  =>  two contiguous 16B loads.
__device__ inline v16h load_a_frag(const half_t* __restrict__ rowp, int g) {
  HFrag f;
  f.q[0] = *(const h8*)(rowp + g * 8);
  f.q[1] = *(const h8*)(rowp + g * 8 + 16);
  return f.v;
}

// B-fragment from N-major ("BT") storage [N][K]: lane col = lane&15,
// half i -> K = (i&7) + (i>>3)*16 + g*8  =>  two contiguous 16B loads.
__device__ inline v16h load_b_frag(const half_t* __restrict__ rowp, int g) {
  HFrag f;
  f.q[0] = *(const h8*)(rowp + g * 8);
  f.q[1] = *(const h8*)(rowp + g * 8 + 16);
  return f.v;
}

// ---------------------------------------------------------------------------
// Elementwise helpers
// ---------------------------------------------------------------------------
__global__ void f2h_kernel(const float* __restrict__ x, half_t* __restrict__ y, size_t n) {
  size_t i = (size_t)blockIdx.x * 256 + threadIdx.x;
  if (i < n) y[i] = (half_t)x[i];
}

// WT[n*K+k] = (half) W[k*N+n]
__global__ void transpose_h_kernel(const float* __restrict__ W, half_t* __restrict__ WT,
                                   int K, int N) {
  size_t i = (size_t)blockIdx.x * 256 + threadIdx.x;
  if (i >= (size_t)K * N) return;
  int n = (int)(i % N);
  int k = (int)(i / N);
  WT[(size_t)n * K + k] = (half_t)W[i];
}

// dec = exp(-0.5*dist);  qsc = dec / sqrt(HD)  (folded into the q projection)
__global__ void decay_kernel(const float* __restrict__ dist, float* __restrict__ dec,
                             float* __restrict__ qsc, int n) {
  int i = blockIdx.x * 256 + threadIdx.x;
  if (i < n) {
    float d = __expf(-0.5f * dist[i]);
    dec[i] = d;
    qsc[i] = d * 0.08838834764831845f;  // 1/sqrt(128)
  }
}

// rq = rh @ Wrq  (tiny GEMM, [B*L,32]@[32,32])
__global__ void rproj_kernel(const float* __restrict__ rh, const float* __restrict__ W,
                             float* __restrict__ out, int total) {
  int idx = blockIdx.x * 256 + threadIdx.x;
  if (idx >= total) return;
  int n  = idx & (NRL - 1);
  int bl = idx >> 5;
  const float* r = rh + (size_t)bl * NRL;
  float s = 0.f;
#pragma unroll
  for (int k = 0; k < NRL; ++k) s += r[k] * W[k * NRL + n];
  out[idx] = s;
}

// rbias[b,l,m] = dec[b,l] * (ALPHA/sqrt(RL)) * dot(rq[b,l], rk[b,m])
__global__ void rbias_kernel(const float* __restrict__ rq, const float* __restrict__ rk,
                             const float* __restrict__ dec, float* __restrict__ rb) {
  size_t idx = (size_t)blockIdx.x * 256 + threadIdx.x;
  const size_t total = (size_t)NB * NL * NL;
  if (idx >= total) return;
  int    m = (int)(idx % NL);
  size_t t = idx / NL;
  int    l = (int)(t % NL);
  int    b = (int)(t / NL);
  const float* a = rq + ((size_t)b * NL + l) * NRL;
  const float* c = rk + ((size_t)b * NL + m) * NRL;
  float s = 0.f;
#pragma unroll
  for (int d = 0; d < NRL; ++d) s += a[d] * c[d];
  rb[idx] = s * 0.08838834764831845f * dec[(size_t)b * NL + l];  // 0.5/sqrt(32) * dec
}

// ---------------------------------------------------------------------------
// WMMA GEMM:  out[M,N] = act(rowscale*(A[M,K] @ BT[N,K]^T) + bias)
// block = 256 threads = 8 waves; block tile 128(M) x 64(N); K-step 32.
// The 64x32 B panel (shared by all 8 waves) is staged once per block into
// double-buffered LDS via CDNA5 async global->LDS DMA (ASYNCcnt), so the
// copy for step k+1 overlaps the WMMAs of step k.
// OUT_MODE: 0 = f32 row-major, 1 = f16 row-major,
//           2 = f16 [B,H,L,HD] (per-head q/k), 3 = f16 [B,H,HD,L] (v^T)
// ---------------------------------------------------------------------------
template <int OUT_MODE, bool LEAKY, bool HAS_BIAS, bool ROWSCALE>
__global__ __launch_bounds__(256) void gemm_wmma(
    const half_t* __restrict__ A, const half_t* __restrict__ BT,
    const float* __restrict__ bias, const float* __restrict__ rowscale,
    float* __restrict__ outF, half_t* __restrict__ outH,
    int M, int N, int K) {
  __shared__ __align__(16) half_t Bs[2][64 * 32];

  const int tid  = threadIdx.x;
  const int lane = tid & 31;
  const int wave = tid >> 5;
  const int ln   = lane & 15;
  const int g    = lane >> 4;
  const int mb   = (blockIdx.y * 8 + wave) * 16;
  const int nb   = blockIdx.x * 64;

  // staging map: thread copies one 16B chunk of the 64x32 panel
  const int srow = tid >> 2;        // 0..63
  const int schk = (tid & 3) * 8;   // half offset 0/8/16/24
  const half_t* sgsrc = BT + (size_t)(nb + srow) * K + schk;

  const half_t* arow = A + (size_t)(mb + ln) * K;

  v8f acc[4];
#pragma unroll
  for (int f = 0; f < 4; ++f) acc[f] = vzero8();

  async_copy_b128(sgsrc, &Bs[0][srow * 32 + schk]);
  wait_async_lds();
  __syncthreads();

  int buf = 0;
  for (int kb = 0; kb < K; kb += 32) {
    const bool more = (kb + 32) < K;
    if (more)  // async DMA of the next panel overlaps this step's WMMAs
      async_copy_b128(sgsrc + kb + 32, &Bs[buf ^ 1][srow * 32 + schk]);
    __builtin_prefetch(arow + kb + 128, 0, 1);

    v16h a = load_a_frag(arow + kb, g);
    const half_t* bbase = &Bs[buf][0];
    v16h bfr[4];
#pragma unroll
    for (int f = 0; f < 4; ++f)
      bfr[f] = load_b_frag(bbase + (size_t)(16 * f + ln) * 32, g);
#pragma unroll
    for (int f = 0; f < 4; ++f)
      acc[f] = __builtin_amdgcn_wmma_f32_16x16x32_f16(false, a, false, bfr[f],
                                                      (short)0, acc[f], false, false);
    if (more) wait_async_lds();
    __syncthreads();
    buf ^= 1;
  }

#pragma unroll
  for (int f = 0; f < 4; ++f) {
#pragma unroll
    for (int j = 0; j < 8; ++j) {
      const int row = mb + j + 8 * g;
      const int col = nb + 16 * f + ln;
      float v = acc[f][j];
      if (ROWSCALE) v *= rowscale[row];
      if (HAS_BIAS) v += bias[col];
      if (LEAKY)    v = v >= 0.f ? v : 0.2f * v;
      if (OUT_MODE == 0) {
        outF[(size_t)row * N + col] = v;
      } else if (OUT_MODE == 1) {
        outH[(size_t)row * N + col] = (half_t)v;
      } else {
        const int b_ = row / NL, l = row % NL;
        const int hh = col / NHD, hd = col % NHD;
        if (OUT_MODE == 2)
          outH[(((size_t)b_ * NH + hh) * NL + l) * NHD + hd] = (half_t)v;
        else
          outH[(((size_t)b_ * NH + hh) * NHD + hd) * NL + l] = (half_t)v;
      }
    }
  }
}

// ---------------------------------------------------------------------------
// Flash-style attention. block = 128 threads = 4 waves; each wave owns a
// 16-row query tile of one (b,h), streams KV in 32-wide chunks.
// q was pre-scaled by dec/sqrt(HD); rbias already carries dec, so:
//   score = qk + rbias;  mask==0 -> -1e-12 (faithful: near zero, NOT -inf)
// ---------------------------------------------------------------------------
__global__ __launch_bounds__(128) void attn_wmma(
    const half_t* __restrict__ qh, const half_t* __restrict__ kh,
    const half_t* __restrict__ vT, const float* __restrict__ rbias,
    const int* __restrict__ mask, half_t* __restrict__ ctxb) {
  __shared__ __align__(16) half_t Pt[4][16 * 32];

  const int lane = threadIdx.x & 31;
  const int wave = threadIdx.x >> 5;
  const int ln   = lane & 15;
  const int g    = lane >> 4;
  const int bh   = blockIdx.y;
  const int b    = bh / NH;
  const int hh   = bh % NH;
  const int qb   = blockIdx.x * 64 + wave * 16;

  const half_t* qrow = qh + ((size_t)bh * NL + qb + ln) * NHD;
  v16h qa[4];
#pragma unroll
  for (int c = 0; c < 4; ++c) qa[c] = load_a_frag(qrow + 32 * c, g);

  float mrun[8], zrun[8];
  v8f   o[8];
#pragma unroll
  for (int j = 0; j < 8; ++j) { mrun[j] = -1e30f; zrun[j] = 0.f; }
#pragma unroll
  for (int n = 0; n < 8; ++n) o[n] = vzero8();

  const half_t* kbase = kh + (size_t)bh * NL * NHD;
  const half_t* vbase = vT + (size_t)bh * NHD * NL;
  const float*  rb    = rbias + ((size_t)b * NL + qb) * NL;
  const int*    mk    = mask + ((size_t)b * NL + qb) * NL;
  half_t*       P     = Pt[wave];

  for (int m0 = 0; m0 < NL; m0 += 32) {
    // prefetch per-element bias/mask for this chunk (issues early)
    float rbv0[8], rbv1[8];
    int   mkv0[8], mkv1[8];
#pragma unroll
    for (int j = 0; j < 8; ++j) {
      const size_t rown = (size_t)(j + 8 * g) * NL + m0 + ln;
      rbv0[j] = rb[rown];
      rbv1[j] = rb[rown + 16];
      mkv0[j] = mk[rown];
      mkv1[j] = mk[rown + 16];
    }

    // batched K-fragment loads, then the 8 score WMMAs
    v16h kf[2][4];
#pragma unroll
    for (int t = 0; t < 2; ++t) {
      const half_t* krow = kbase + (size_t)(m0 + 16 * t + ln) * NHD;
#pragma unroll
      for (int c = 0; c < 4; ++c) kf[t][c] = load_b_frag(krow + 32 * c, g);
    }
    v8f s[2];
    s[0] = vzero8();
    s[1] = vzero8();
#pragma unroll
    for (int t = 0; t < 2; ++t)
#pragma unroll
      for (int c = 0; c < 4; ++c)
        s[t] = __builtin_amdgcn_wmma_f32_16x16x32_f16(false, qa[c], false, kf[t][c],
                                                      (short)0, s[t], false, false);

    // issue V fragments now; they cover the softmax VALU latency
    v16h vf[8];
#pragma unroll
    for (int n = 0; n < 8; ++n)
      vf[n] = load_b_frag(vbase + (size_t)(16 * n + ln) * NL + m0, g);

    // streaming softmax over this 32-wide chunk
    float fj[8];
#pragma unroll
    for (int j = 0; j < 8; ++j) {
      const int row = j + 8 * g;
      float v0 = s[0][j] + rbv0[j];
      float v1 = s[1][j] + rbv1[j];
      if (mkv0[j] == 0) v0 = -1e-12f;
      if (mkv1[j] == 0) v1 = -1e-12f;

      float rm = fmaxf(v0, v1);
#pragma unroll
      for (int off = 1; off < 16; off <<= 1) rm = fmaxf(rm, __shfl_xor(rm, off, 32));
      const float mnew = fmaxf(mrun[j], rm);
      const float sc   = __expf(mrun[j] - mnew);
      mrun[j] = mnew;
      v0 = __expf(v0 - mnew);
      v1 = __expf(v1 - mnew);
      float rs = v0 + v1;
#pragma unroll
      for (int off = 1; off < 16; off <<= 1) rs += __shfl_xor(rs, off, 32);
      zrun[j] = zrun[j] * sc + rs;
      fj[j]   = sc;

      // spill P tile (C-layout) to per-wave LDS for the C->A transpose
      P[(size_t)row * 32 + ln]      = (half_t)v0;
      P[(size_t)row * 32 + 16 + ln] = (half_t)v1;
    }

    // rescale running output
#pragma unroll
    for (int n = 0; n < 8; ++n) {
#pragma unroll
      for (int j = 0; j < 8; ++j) o[n][j] *= fj[j];
    }

    // reload P in A-layout, then the 8 P@V WMMAs (vf already in flight)
    v16h pf = load_a_frag(P + (size_t)ln * 32, g);
#pragma unroll
    for (int n = 0; n < 8; ++n)
      o[n] = __builtin_amdgcn_wmma_f32_16x16x32_f16(false, pf, false, vf[n],
                                                    (short)0, o[n], false, false);
  }

  // finalize: ctx row-major [B*L, D] with col = h*HD + hd (f16 for Wo GEMM)
  float rz[8];
#pragma unroll
  for (int j = 0; j < 8; ++j) rz[j] = 1.f / zrun[j];
#pragma unroll
  for (int n = 0; n < 8; ++n) {
#pragma unroll
    for (int j = 0; j < 8; ++j) {
      const int row = qb + j + 8 * g;
      ctxb[((size_t)b * NL + row) * ND + hh * NHD + 16 * n + ln] =
          (half_t)(o[n][j] * rz[j]);
    }
  }
}

// ---------------------------------------------------------------------------
// LayerNorm over last dim of (xa + xb); optional f32 and f16 outputs.
// ---------------------------------------------------------------------------
__global__ __launch_bounds__(256) void ln_kernel(
    const float* __restrict__ xa, const float* __restrict__ xb,
    const float* __restrict__ gam, const float* __restrict__ bet,
    float* __restrict__ outF, half_t* __restrict__ outH) {
  const int row = blockIdx.x;
  const float* p = xa + (size_t)row * ND;
  const float* q = xb + (size_t)row * ND;

  float s = 0.f, ss = 0.f;
  for (int i = threadIdx.x; i < ND; i += 256) {
    float v = p[i] + q[i];
    s += v;
    ss += v * v;
  }
#pragma unroll
  for (int off = 16; off; off >>= 1) {
    s  += __shfl_xor(s, off, 32);
    ss += __shfl_xor(ss, off, 32);
  }
  __shared__ float sh[2][8];
  const int lane = threadIdx.x & 31, w = threadIdx.x >> 5;
  if (lane == 0) { sh[0][w] = s; sh[1][w] = ss; }
  __syncthreads();
  if (w == 0) {
    s  = (lane < 8) ? sh[0][lane] : 0.f;
    ss = (lane < 8) ? sh[1][lane] : 0.f;
#pragma unroll
    for (int off = 4; off; off >>= 1) {
      s  += __shfl_xor(s, off, 32);
      ss += __shfl_xor(ss, off, 32);
    }
    if (lane == 0) { sh[0][0] = s; sh[1][0] = ss; }
  }
  __syncthreads();
  const float mean = sh[0][0] * (1.f / ND);
  const float var  = sh[1][0] * (1.f / ND) - mean * mean;
  const float inv  = rsqrtf(var + 1e-5f);
  for (int i = threadIdx.x; i < ND; i += 256) {
    float v = (p[i] + q[i] - mean) * inv * gam[i] + bet[i];
    if (outF) outF[(size_t)row * ND + i] = v;
    if (outH) outH[(size_t)row * ND + i] = (half_t)v;
  }
}

// ---------------------------------------------------------------------------
// Host orchestration
// ---------------------------------------------------------------------------
static inline int blocks_for(size_t n) { return (int)((n + 255) / 256); }

extern "C" void kernel_launch(void* const* d_in, const int* in_sizes, int n_in,
                              void* d_out, int out_size, void* d_ws, size_t ws_size,
                              hipStream_t stream) {
  (void)in_sizes; (void)n_in; (void)out_size; (void)ws_size;

  const float* h    = (const float*)d_in[0];
  const float* dist = (const float*)d_in[1];
  const float* rh   = (const float*)d_in[2];
  const int*   mask = (const int*)d_in[3];
  const float* Wq   = (const float*)d_in[4];
  const float* Wk   = (const float*)d_in[5];
  const float* Wv   = (const float*)d_in[6];
  const float* Wrq  = (const float*)d_in[7];
  const float* Wrk  = (const float*)d_in[8];
  const float* Wo   = (const float*)d_in[9];
  const float* W1   = (const float*)d_in[10];
  const float* b1   = (const float*)d_in[11];
  const float* W2   = (const float*)d_in[12];
  const float* b2   = (const float*)d_in[13];
  const float* g1   = (const float*)d_in[14];
  const float* be1  = (const float*)d_in[15];
  const float* g2   = (const float*)d_in[16];
  const float* be2  = (const float*)d_in[17];

  char*  ws  = (char*)d_ws;
  size_t off = 0;
  auto take = [&](size_t bytes) -> char* {
    char* p = ws + off;
    off += (bytes + 255) & ~(size_t)255;
    return p;
  };

  const size_t MROWS = (size_t)NB * NL;            // 4096
  half_t* hb   = (half_t*)take(MROWS * ND * 2);
  half_t* WqT  = (half_t*)take((size_t)ND * ND * 2);
  half_t* WkT  = (half_t*)take((size_t)ND * ND * 2);
  half_t* WvT  = (half_t*)take((size_t)ND * ND * 2);
  half_t* WoT  = (half_t*)take((size_t)ND * ND * 2);
  half_t* W1T  = (half_t*)take((size_t)ND * ND * 2);
  half_t* W2T  = (half_t*)take((size_t)ND * ND * 2);
  half_t* qhd  = (half_t*)take(MROWS * ND * 2);    // [B,H,L,HD]
  half_t* khd  = (half_t*)take(MROWS * ND * 2);    // [B,H,L,HD]
  half_t* vTd  = (half_t*)take(MROWS * ND * 2);    // [B,H,HD,L]
  half_t* ctxb = (half_t*)take(MROWS * ND * 2);
  half_t* h1b  = (half_t*)take(MROWS * ND * 2);
  half_t* t1b  = (half_t*)take(MROWS * ND * 2);
  float*  rq   = (float*)take(MROWS * NRL * 4);
  float*  rk   = (float*)take(MROWS * NRL * 4);
  float*  rbias= (float*)take((size_t)NB * NL * NL * 4);
  float*  dec  = (float*)take(MROWS * 4);
  float*  qsc  = (float*)take(MROWS * 4);
  float*  hsa  = (float*)take(MROWS * ND * 4);
  float*  h1f  = (float*)take(MROWS * ND * 4);
  float*  hf   = (float*)take(MROWS * ND * 4);

  // 1) f32 -> f16 conversions / weight transposes
  f2h_kernel<<<blocks_for(MROWS * ND), 256, 0, stream>>>(h, hb, MROWS * ND);
  const size_t wN = (size_t)ND * ND;
  transpose_h_kernel<<<blocks_for(wN), 256, 0, stream>>>(Wq, WqT, ND, ND);
  transpose_h_kernel<<<blocks_for(wN), 256, 0, stream>>>(Wk, WkT, ND, ND);
  transpose_h_kernel<<<blocks_for(wN), 256, 0, stream>>>(Wv, WvT, ND, ND);
  transpose_h_kernel<<<blocks_for(wN), 256, 0, stream>>>(Wo, WoT, ND, ND);
  transpose_h_kernel<<<blocks_for(wN), 256, 0, stream>>>(W1, W1T, ND, ND);
  transpose_h_kernel<<<blocks_for(wN), 256, 0, stream>>>(W2, W2T, ND, ND);

  // 2) decay/qscale, relative bias (decay folded in)
  decay_kernel<<<blocks_for(MROWS), 256, 0, stream>>>(dist, dec, qsc, (int)MROWS);
  rproj_kernel<<<blocks_for(MROWS * NRL), 256, 0, stream>>>(rh, Wrq, rq, (int)(MROWS * NRL));
  rproj_kernel<<<blocks_for(MROWS * NRL), 256, 0, stream>>>(rh, Wrk, rk, (int)(MROWS * NRL));
  rbias_kernel<<<blocks_for((size_t)NB * NL * NL), 256, 0, stream>>>(rq, rk, dec, rbias);

  // 3) q/k/v projections (WMMA); q carries dec/sqrt(HD) row-scale
  dim3 ggrid(ND / 64, (int)(MROWS / 128));
  gemm_wmma<2, false, false, true><<<ggrid, 256, 0, stream>>>(
      hb, WqT, nullptr, qsc, nullptr, qhd, (int)MROWS, ND, ND);
  gemm_wmma<2, false, false, false><<<ggrid, 256, 0, stream>>>(
      hb, WkT, nullptr, nullptr, nullptr, khd, (int)MROWS, ND, ND);
  gemm_wmma<3, false, false, false><<<ggrid, 256, 0, stream>>>(
      hb, WvT, nullptr, nullptr, nullptr, vTd, (int)MROWS, ND, ND);

  // 4) flash attention (WMMA score + WMMA P@V)
  attn_wmma<<<dim3(NL / 64, NB * NH), 128, 0, stream>>>(qhd, khd, vTd, rbias, mask, ctxb);

  // 5) output projection + leaky relu -> h_sa (f32)
  gemm_wmma<0, true, false, false><<<ggrid, 256, 0, stream>>>(
      ctxb, WoT, nullptr, nullptr, hsa, nullptr, (int)MROWS, ND, ND);

  // 6) h1 = LN(h_sa + h)
  ln_kernel<<<(int)MROWS, 256, 0, stream>>>(hsa, h, g1, be1, h1f, h1b);

  // 7) FFN
  gemm_wmma<1, true, true, false><<<ggrid, 256, 0, stream>>>(
      h1b, W1T, b1, nullptr, nullptr, t1b, (int)MROWS, ND, ND);
  gemm_wmma<0, false, true, false><<<ggrid, 256, 0, stream>>>(
      t1b, W2T, b2, nullptr, hf, nullptr, (int)MROWS, ND, ND);

  // 8) out = LN(h1 + hf)
  ln_kernel<<<(int)MROWS, 256, 0, stream>>>(h1f, hf, g2, be2, (float*)d_out, nullptr);
}
